// MatrixMessage_55705725829531
// MI455X (gfx1250) — matrix operand order repaired and checked by reading, not compiled, not run on hardware
//
#include <hip/hip_runtime.h>

typedef __attribute__((ext_vector_type(16))) __bf16 v16bf;
typedef __attribute__((ext_vector_type(8)))  float  v8f;

#define HIDDEN   128
#define EDGE_DIM 8
#define SUB_E    16           // WMMA M dimension
#define TILE_E   32           // two 16-edge sub-tiles per block iteration
#define WAVES    8            // one 16-col n-tile per wave -> 8*16 = 128 hidden
#define BLOCK    (WAVES * 32)

// ---- dynamic LDS layout -------------------------------------------------
// sB : [8 types][8 ntile][4 kk][32 lane][16 bf16] (B-fragment swizzled) 256 KB
// sA : [2 sub][4 kk][32 lane][16 bf16]            (A-fragment swizzled)   8 KB
// sF : [32 edges][8] float (one-hot weights)                              1 KB
// sMask[2] : per-sub-tile bitmask of edge types present
#define B_ELEMS (EDGE_DIM * 8 * 4 * 32 * 16)
#define B_BYTES (B_ELEMS * 2)                 // 262144
#define A_OFF   B_BYTES
#define A_SUB_ELEMS (4 * 32 * 16)             // 2048 bf16 per sub-tile
#define A_BYTES (2 * A_SUB_ELEMS * 2)         // 8192
#define F_OFF   (A_OFF + A_BYTES)
#define F_BYTES (TILE_E * EDGE_DIM * 4)       // 1024
#define M_OFF   (F_OFF + F_BYTES)
#define SMEM_BYTES (M_OFF + 16)               // 271376 < 320 KB/WGP

__global__ __launch_bounds__(BLOCK, 1)
void MatrixMessage_55705725829531_kernel(const float* __restrict__ h_w,
                                         const float* __restrict__ edge_features,
                                         const float* __restrict__ edge_matrices,
                                         float* __restrict__ out, int nE)
{
    extern __shared__ __align__(32) char smem[];
    __bf16* sB    = (__bf16*)smem;
    __bf16* sA    = (__bf16*)(smem + A_OFF);
    float*  sF    = (float*)(smem + F_OFF);
    int*    sMask = (int*)(smem + M_OFF);

    const int tid  = threadIdx.x;
    const int lane = tid & 31;
    const int wave = tid >> 5;

    // ---- Phase 0: stage all 8 matrices into LDS as bf16 B-fragments.
    // B fragment (32x16), this code's k order: lane -> col n = lane&15,
    // element j -> K = (lane>>4)*16 + j  => 16 K-contiguous f32 per group.
    for (int g = tid; g < B_ELEMS / 16; g += BLOCK) {
        int t     = g >> 10;            // 8*4*32 = 1024 groups per type
        int r     = g & 1023;
        int ntile = r >> 7;
        int r2    = r & 127;
        int kk    = r2 >> 5;
        int ln    = r2 & 31;
        int n     = ntile * 16 + (ln & 15);
        int d0    = kk * 32 + (ln >> 4) * 16;
        const float* src = edge_matrices + ((size_t)(t * HIDDEN + n) * HIDDEN + d0);
        __bf16* dst = sB + (size_t)g * 16;
        #pragma unroll
        for (int j = 0; j < 16; ++j) dst[j] = (__bf16)src[j];
    }
    __syncthreads();

    const int nTiles = (nE + TILE_E - 1) / TILE_E;
    for (int tile = blockIdx.x; tile < nTiles; tile += (int)gridDim.x) {
        const int e0 = tile * TILE_E;
        if (tid < 2) sMask[tid] = 0;
        __syncthreads();

        // ---- stage A: 32 edges x 128 d of h_w -> two bf16 A-fragment tiles.
        // A (16x32 bf16): lane = hi*16 + mrow, element j ->
        // K = hi*16 + j.
        #pragma unroll
        for (int s = 0; s < 2; ++s) {
            int u    = tid + s * BLOCK;     // 512 octet-units: 32 rows x 16 octets
            int m    = u >> 4;              // 0..31 (edge row in tile)
            int d0   = (u & 15) * 8;        // 0..120 step 8 (coalesced 32B reads)
            int e    = e0 + m;
            float v[8];
            if (e < nE) {
                const float* src = h_w + (size_t)e * HIDDEN + d0;
                #pragma unroll
                for (int j = 0; j < 8; ++j) v[j] = src[j];
            } else {
                #pragma unroll
                for (int j = 0; j < 8; ++j) v[j] = 0.f;
            }
            int sub  = m >> 4;
            int mrow = m & 15;
            int kk   = d0 >> 5;
            int K0   = d0 & 31;             // 0,8,16,24
            int hi   = K0 >> 4;
            int jb   = K0 & 8;              // 0 or 8
            __bf16* dst = sA + sub * A_SUB_ELEMS
                             + ((kk * 32 + hi * 16 + mrow) * 16 + jb);
            #pragma unroll
            for (int j = 0; j < 8; ++j) dst[j] = (__bf16)v[j];
        }
        // ---- stage one-hot weights + per-sub-tile presence bitmasks
        if (tid < TILE_E * EDGE_DIM) {
            int e = tid >> 3, k = tid & 7;
            float f = (e0 + e < nE)
                ? edge_features[(size_t)(e0 + e) * EDGE_DIM + k] : 0.f;
            sF[tid] = f;
            if (f != 0.f) atomicOr(&sMask[e >> 4], 1 << k);
        }
        __syncthreads();

        // ---- compute: wave w -> rows e0..e0+31, cols [w*16, w*16+16)
        const int mrow = lane & 15;
        float fr0[EDGE_DIM], fr1[EDGE_DIM];
        #pragma unroll
        for (int k = 0; k < EDGE_DIM; ++k) {
            fr0[k] = sF[mrow * EDGE_DIM + k];
            fr1[k] = sF[(16 + mrow) * EDGE_DIM + k];
        }
        const int tmask0 = sMask[0];
        const int tmask1 = sMask[1];

        v16bf afrag0[4], afrag1[4];
        #pragma unroll
        for (int kk = 0; kk < 4; ++kk) {
            afrag0[kk] = *(const v16bf*)(sA + (kk * 32 + lane) * 16);
            afrag1[kk] = *(const v16bf*)(sA + A_SUB_ELEMS + (kk * 32 + lane) * 16);
        }

        v8f acc0 = {}, acc1 = {};
        for (int k = 0; k < EDGE_DIM; ++k) {
            const bool p0 = (tmask0 >> k) & 1;   // wave-uniform: EXEC stays full
            const bool p1 = (tmask1 >> k) & 1;
            if (!(p0 | p1)) continue;
            __bf16 fb0 = (__bf16)fr0[k];
            __bf16 fb1 = (__bf16)fr1[k];
            v16bf fs0, fs1;
            #pragma unroll
            for (int j = 0; j < 16; ++j) { fs0[j] = fb0; fs1[j] = fb1; }
            const __bf16* bbase = sB + (size_t)((k * 8 + wave) * 4) * 32 * 16;
            #pragma unroll
            for (int kk = 0; kk < 4; ++kk) {
                // one B fragment feeds two independent accumulator chains
                v16bf b = *(const v16bf*)(bbase + (kk * 32 + lane) * 16);
                if (p0) {
                    v16bf a0 = afrag0[kk] * fs0;  // rows scaled by one-hot weight
                    acc0 = __builtin_amdgcn_wmma_f32_16x16x32_bf16(
                            false, a0, false, b, (short)0, acc0, false, false);
                }
                if (p1) {
                    v16bf a1 = afrag1[kk] * fs1;
                    acc1 = __builtin_amdgcn_wmma_f32_16x16x32_bf16(
                            false, a1, false, b, (short)0, acc1, false, false);
                }
            }
        }

        // ---- store D tiles (16x16 f32): VGPR r -> row (lane>>4)*8+r, col lane&15
        {
            const int col   = lane & 15;
            const int mbase = (lane >> 4) * 8;
            #pragma unroll
            for (int r = 0; r < 8; ++r) {
                int e = e0 + mbase + r;
                if (e < nE)
                    out[(size_t)e * HIDDEN + wave * 16 + col] = acc0[r];
                int e1 = e + 16;
                if (e1 < nE)
                    out[(size_t)e1 * HIDDEN + wave * 16 + col] = acc1[r];
            }
        }
        // prefetch next tile's h_w slab (global_prefetch_b8)
        {
            int nt = tile + (int)gridDim.x;
            if (nt < nTiles) {
                const char* p = (const char*)(h_w + (size_t)nt * TILE_E * HIDDEN);
                __builtin_prefetch(p + tid * 64, 0, 1);
            }
        }
        __syncthreads();
    }
}

extern "C" void kernel_launch(void* const* d_in, const int* in_sizes, int n_in,
                              void* d_out, int out_size, void* d_ws, size_t ws_size,
                              hipStream_t stream) {
    // inputs: 0=h_v (unused), 1=h_w, 2=edge_features, 3=edge_matrices
    const float* h_w = (const float*)d_in[1];
    const float* ef  = (const float*)d_in[2];
    const float* em  = (const float*)d_in[3];
    float* out = (float*)d_out;

    const int nE     = in_sizes[1] / HIDDEN;
    const int nTiles = (nE + TILE_E - 1) / TILE_E;
    const int grid   = nTiles < 512 ? nTiles : 512;  // persistent blocks; matrix
                                                     // staging amortized ~20x

    static_assert(SMEM_BYTES <= 320 * 1024, "fits CDNA5 WGP LDS");
    (void)hipFuncSetAttribute((const void*)MatrixMessage_55705725829531_kernel,
                              hipFuncAttributeMaxDynamicSharedMemorySize,
                              SMEM_BYTES);
    MatrixMessage_55705725829531_kernel<<<grid, BLOCK, SMEM_BYTES, stream>>>(
        h_w, ef, em, out, nE);
}
